// ColorNetwork_59837484367921
// MI455X (gfx1250) — matrix operand order repaired and checked by reading, not compiled, not run on hardware
//
#include <hip/hip_runtime.h>

typedef __attribute__((ext_vector_type(16))) _Float16 v16h;
typedef __attribute__((ext_vector_type(8)))  float    v8f;
typedef unsigned int u32;

#define H_LD  66   // padded row stride (halves) for 64-wide h rows: 132B -> bank step 33 (coprime 64)
#define W_LD0 34   // padded K stride for w0 (K=32): 68B -> bank step 17
#define W_LD1 66   // padded K stride for w1/w2 (K=64)

union V16 { v16h v; u32 u[8]; _Float16 h[16]; };

// A-fragment (16x32 f16, MxK): lane m (0-15) = row m; lanes 16-31 same rows, K+8.
// VGPR j<4 holds K=2j,2j+1 (+hs); VGPR j>=4 holds K=16+2(j-4),+1 (+hs).
__device__ __forceinline__ v16h load_a_frag(const _Float16* s, int ld, int rowbase,
                                            int lane, int kofs) {
  const int m  = lane & 15;
  const int hs = (lane >> 4) << 3;
  const u32* r32 = (const u32*)(s + (rowbase + m) * ld + kofs + hs);
  V16 a;
#pragma unroll
  for (int j = 0; j < 8; ++j) a.u[j] = r32[(j < 4) ? j : (j + 4)];
  return a.v;
}

// B-fragment (32x16 f16, KxN): lane n (0-15) = col n; lanes 16-31 same cols, K+8 (same K order as A).
// Weights staged transposed wt[n][k] so each VGPR pair-load is one 32-bit LDS load.
__device__ __forceinline__ v16h load_b_frag(const _Float16* wt, int ld, int n0,
                                            int lane, int kofs) {
  const int n    = lane & 15;
  const int hs16 = (lane >> 4) << 3;
  const u32* c32 = (const u32*)(wt + (n0 + n) * ld + kofs + hs16);
  V16 b;
#pragma unroll
  for (int j = 0; j < 8; ++j) b.u[j] = c32[(j < 4) ? j : (j + 4)];
  return b.v;
}

// C/D layout: VGPR r, lanes 0-15 = (M=r, N=lane); lanes 16-31 = (M=8+r, N=lane-16).
__device__ __forceinline__ v8f bias_c(const float* sb, int n0, int lane) {
  float bv = sb[n0 + (lane & 15)];
  v8f c;
#pragma unroll
  for (int r = 0; r < 8; ++r) c[r] = bv;
  return c;
}

__device__ __forceinline__ void store_relu(_Float16* s, int ld, int rowbase, int n0,
                                           int lane, v8f c) {
  const int n  = lane & 15;
  const int mo = (lane >> 4) << 3;
  _Float16* dst = s + (rowbase + mo) * ld + n0 + n;
#pragma unroll
  for (int r = 0; r < 8; ++r) dst[r * ld] = (_Float16)fmaxf(c[r], 0.0f);
}

__global__ __launch_bounds__(256)
void colornet_wmma_kernel(const float* __restrict__ coords,
                          const float* __restrict__ g0, const float* __restrict__ g1,
                          const float* __restrict__ g2, const float* __restrict__ g3,
                          const float* __restrict__ g4, const float* __restrict__ g5,
                          const float* __restrict__ g6, const float* __restrict__ g7,
                          const float* __restrict__ lpe,
                          const float* __restrict__ w0, const float* __restrict__ b0,
                          const float* __restrict__ w1, const float* __restrict__ b1,
                          const float* __restrict__ w2, const float* __restrict__ b2,
                          float* __restrict__ out, int nPoints) {
  __shared__ __align__(16) _Float16 s_h[256 * H_LD];      // feat rows (cols 0-31), then h
  __shared__ __align__(16) _Float16 s_w0t[64 * W_LD0];    // w0^T f16, K padded 26->32
  __shared__ __align__(16) _Float16 s_w1t[64 * W_LD1];    // w1^T f16
  __shared__ __align__(16) _Float16 s_w2t[16 * W_LD1];    // w2^T f16, N padded 4->16
  __shared__ float s_b0[64], s_b1[64], s_b2[16];

  const int tid = threadIdx.x;

  // ---- stage weights (transposed, f16) cooperatively ----
  for (int i = tid; i < 64 * 32; i += 256) {
    int n = i >> 5, k = i & 31;
    s_w0t[n * W_LD0 + k] = (_Float16)((k < 26) ? w0[k * 64 + n] : 0.0f);
  }
  for (int i = tid; i < 64 * 64; i += 256) {
    int n = i >> 6, k = i & 63;
    s_w1t[n * W_LD1 + k] = (_Float16)w1[k * 64 + n];
  }
  for (int i = tid; i < 16 * 64; i += 256) {
    int n = i >> 6, k = i & 63;
    s_w2t[n * W_LD1 + k] = (_Float16)((n < 4) ? w2[k * 4 + n] : 0.0f);
  }
  if (tid < 64) { s_b0[tid] = b0[tid]; s_b1[tid] = b1[tid]; }
  if (tid < 16) s_b2[tid] = (tid < 4) ? b2[tid] : 0.0f;

  // ---- front-end: one point per thread ----
  const int pBase = blockIdx.x * 256;
  const int pc = min(pBase + tid, nPoints - 1);
  const float u = fminf(fmaxf(coords[2 * pc + 0], 0.0f), 1.0f - 1e-6f);
  const float v = fminf(fmaxf(coords[2 * pc + 1], 0.0f), 1.0f - 1e-6f);

  float e0 = 0.0f, e1 = 0.0f;
  const float* gp[8] = {g0, g1, g2, g3, g4, g5, g6, g7};
#pragma unroll
  for (int l = 0; l < 8; ++l) {
    const int r = 16 << l;
    const float scale = (float)(r - 1);
    float fu = u * scale, fv = v * scale;
    int i0 = min(max((int)floorf(fu), 0), r - 2);
    int j0 = min(max((int)floorf(fv), 0), r - 2);
    float lu = fu - (float)i0, lv = fv - (float)j0;
    const float2* p00 = (const float2*)(gp[l] + (size_t)(j0 * r + i0) * 2);
    float2 f00 = p00[0], f10 = p00[1];
    const float2* p01 = p00 + r;
    float2 f01 = p01[0], f11 = p01[1];
    float w00 = (1.0f - lu) * (1.0f - lv), w10 = lu * (1.0f - lv);
    float w01 = (1.0f - lu) * lv,          w11 = lu * lv;
    e0 += f00.x * w00 + f10.x * w10 + f01.x * w01 + f11.x * w11;
    e1 += f00.y * w00 + f10.y * w10 + f01.y * w01 + f11.y * w11;
  }

  // LPE gather: 24 floats/corner = 6 x float4 (records are 96B, 16B aligned)
  const float fuL = u * 128.0f, fvL = v * 128.0f;
  const int i0L = min(max((int)floorf(fuL), 0), 127);
  const int j0L = min(max((int)floorf(fvL), 0), 127);
  const float lu = fuL - (float)i0L, lv = fvL - (float)j0L;
  {
    const float wl00 = (1.0f - lu) * (1.0f - lv), wl10 = lu * (1.0f - lv);
    const float wl01 = (1.0f - lu) * lv,          wl11 = lu * lv;
    const float4* c00 = (const float4*)(lpe + (size_t)(j0L * 129 + i0L) * 24);
    const float4* c10 = c00 + 6;
    const float4* c01 = (const float4*)((const float*)c00 + (size_t)129 * 24);
    const float4* c11 = c01 + 6;
    float coeff[24];
#pragma unroll
    for (int q = 0; q < 6; ++q) {
      float4 a = c00[q], b = c10[q], c = c01[q], d = c11[q];
      coeff[4 * q + 0] = a.x * wl00 + b.x * wl10 + c.x * wl01 + d.x * wl11;
      coeff[4 * q + 1] = a.y * wl00 + b.y * wl10 + c.y * wl01 + d.y * wl11;
      coeff[4 * q + 2] = a.z * wl00 + b.z * wl10 + c.z * wl01 + d.z * wl11;
      coeff[4 * q + 3] = a.w * wl00 + b.w * wl10 + c.w * wl01 + d.w * wl11;
    }
    // feat row: [enc(2), base(8), gate*pe(16)], padded to 32 with zeros
    _Float16* frow = s_h + tid * H_LD;
    frow[0] = (_Float16)e0;
    frow[1] = (_Float16)e1;
#pragma unroll
    for (int k = 0; k < 8; ++k) frow[2 + k] = (_Float16)coeff[k];
    const float TWO_PI = 6.283185307179586f;
#pragma unroll
    for (int j = 0; j < 4; ++j) {
      float fr = TWO_PI * (float)(1 << j);
      float uu = lu * fr, vv = lv * fr;
      frow[10 + j] = (_Float16)(coeff[8 + j]  * __cosf(uu));
      frow[14 + j] = (_Float16)(coeff[12 + j] * __sinf(uu));
      frow[18 + j] = (_Float16)(coeff[16 + j] * __cosf(vv));
      frow[22 + j] = (_Float16)(coeff[20 + j] * __sinf(vv));
    }
#pragma unroll
    for (int k = 26; k < 32; ++k) frow[k] = (_Float16)0.0f;
  }

  __syncthreads();

  // ---- MLP: each wave handles its own 32 rows as two 16-row WMMA tiles ----
  const int wave = tid >> 5;
  const int lane = tid & 31;
#pragma unroll
  for (int mt = 0; mt < 2; ++mt) {
    const int rb = wave * 32 + mt * 16;

    // layer 0: 16x32 feat x 32x64 w0 (a0 fully loaded before stores overwrite)
    v16h a0 = load_a_frag(s_h, H_LD, rb, lane, 0);
#pragma unroll
    for (int nt = 0; nt < 4; ++nt) {
      v16h bf = load_b_frag(s_w0t, W_LD0, nt * 16, lane, 0);
      v8f c = bias_c(s_b0, nt * 16, lane);
      c = __builtin_amdgcn_wmma_f32_16x16x32_f16(false, a0, false, bf,
                                                 (short)0, c, false, false);
      store_relu(s_h, H_LD, rb, nt * 16, lane, c);
    }

    // layer 1: 16x64 h x 64x64 w1 (K split 2x32; A regs loaded before overwrite)
    v16h a1lo = load_a_frag(s_h, H_LD, rb, lane, 0);
    v16h a1hi = load_a_frag(s_h, H_LD, rb, lane, 32);
#pragma unroll
    for (int nt = 0; nt < 4; ++nt) {
      v16h blo = load_b_frag(s_w1t, W_LD1, nt * 16, lane, 0);
      v16h bhi = load_b_frag(s_w1t, W_LD1, nt * 16, lane, 32);
      v8f c = bias_c(s_b1, nt * 16, lane);
      c = __builtin_amdgcn_wmma_f32_16x16x32_f16(false, a1lo, false, blo,
                                                 (short)0, c, false, false);
      c = __builtin_amdgcn_wmma_f32_16x16x32_f16(false, a1hi, false, bhi,
                                                 (short)0, c, false, false);
      store_relu(s_h, H_LD, rb, nt * 16, lane, c);
    }

    // layer 2: 16x64 h x 64x16 w2 (N padded 4->16), sigmoid, store cols 0-3
    v16h a2lo = load_a_frag(s_h, H_LD, rb, lane, 0);
    v16h a2hi = load_a_frag(s_h, H_LD, rb, lane, 32);
    v16h b2lo = load_b_frag(s_w2t, W_LD1, 0, lane, 0);
    v16h b2hi = load_b_frag(s_w2t, W_LD1, 0, lane, 32);
    v8f c = bias_c(s_b2, 0, lane);
    c = __builtin_amdgcn_wmma_f32_16x16x32_f16(false, a2lo, false, b2lo,
                                               (short)0, c, false, false);
    c = __builtin_amdgcn_wmma_f32_16x16x32_f16(false, a2hi, false, b2hi,
                                               (short)0, c, false, false);
    const int n  = lane & 15;
    const int mo = (lane >> 4) << 3;
    if (n < 4) {
#pragma unroll
      for (int r = 0; r < 8; ++r) {
        int pt = pBase + rb + mo + r;
        if (pt < nPoints)
          out[(size_t)pt * 4 + n] = 1.0f / (1.0f + __expf(-c[r]));
      }
    }
  }
}

extern "C" void kernel_launch(void* const* d_in, const int* in_sizes, int n_in,
                              void* d_out, int out_size, void* d_ws, size_t ws_size,
                              hipStream_t stream) {
  const float* coords = (const float*)d_in[0];
  const float* g0 = (const float*)d_in[1];
  const float* g1 = (const float*)d_in[2];
  const float* g2 = (const float*)d_in[3];
  const float* g3 = (const float*)d_in[4];
  const float* g4 = (const float*)d_in[5];
  const float* g5 = (const float*)d_in[6];
  const float* g6 = (const float*)d_in[7];
  const float* g7 = (const float*)d_in[8];
  const float* lpe = (const float*)d_in[9];
  const float* w0 = (const float*)d_in[10];
  const float* b0 = (const float*)d_in[11];
  const float* w1 = (const float*)d_in[12];
  const float* b1 = (const float*)d_in[13];
  const float* w2 = (const float*)d_in[14];
  const float* b2 = (const float*)d_in[15];
  float* out = (float*)d_out;

  const int nPoints = in_sizes[0] / 2;
  const int blocks = (nPoints + 255) / 256;
  colornet_wmma_kernel<<<blocks, 256, 0, stream>>>(
      coords, g0, g1, g2, g3, g4, g5, g6, g7, lpe,
      w0, b0, w1, b1, w2, b2, out, nPoints);
}
